// LSTMModel_21526376087636
// MI455X (gfx1250) — matrix operand order repaired and checked by reading, not compiled, not run on hardware
//
#include <hip/hip_runtime.h>

// ---------------------------------------------------------------------------
// CDNA5 (gfx1250) bi-LSTM, wave32, WMMA bf16 16x16x32 with f32 accumulate.
// One block = 16 batch rows x 4 waves; wave w owns unit range [16w,16w+16)
// and carries 4 accumulators (one per gate quadrant i/f/g/o), so the LSTM
// cell update is fully in-register. h lives in LDS as the next step's
// A-fragment; c lives in a v8f register. 2 barriers per timestep.
// ---------------------------------------------------------------------------

typedef __attribute__((ext_vector_type(16))) __bf16 v16bf;
typedef __attribute__((ext_vector_type(8)))  __bf16 v8bf;
typedef __attribute__((ext_vector_type(8)))  float  v8f;

#define DEV __device__ __forceinline__

DEV unsigned short f2bfbits(float f) {
  unsigned u = __builtin_bit_cast(unsigned, f);
  u += 0x7FFFu + ((u >> 16) & 1u);           // round-to-nearest-even
  return (unsigned short)(u >> 16);
}

DEV float sigmoidf_(float x) { return 1.0f / (1.0f + __expf(-x)); }

// A-fragment load (16x32 bf16, MxK) per ISA 7.12.2:
//  lanes 0-15: M=lane, halves 0-7 => K=0..7, halves 8-15 => K=16..23
//  lanes 16-31: M=lane-16, K offset +8 (and +8 in the upper half)
DEV v16bf load_A_frag(const unsigned short* sm, int ldK, int kc) {
  const int lane = threadIdx.x & 31;
  const int M = lane & 15;
  const int ko = (lane >> 4) << 3;                 // 0 or 8
  const unsigned short* base = sm + M * ldK + kc * 32;
  v8bf lo = *(const v8bf*)(base + ko);             // K = ko .. ko+7
  v8bf hi = *(const v8bf*)(base + 16 + ko);        // K = 16+ko .. 16+ko+7
  return __builtin_shufflevector(lo, hi, 0,1,2,3,4,5,6,7,8,9,10,11,12,13,14,15);
}

// B-fragment load (32x16 bf16, KxN), packed so each lane reads 32 contiguous
// bytes: packed[(tile*kch + kc)*512 + lane*16 + h]
DEV v16bf load_B_frag(const unsigned short* g) {
  const int lane = threadIdx.x & 31;
  return *(const v16bf*)(g + lane * 16);
}

// Pack W[256][Kin] (fp32, row = gate unit N, col = K) into bf16 B-fragments.
// B[k][n] = W[n][k]; K >= Kin zero-padded. Lane L half h -> K = (L>=16?8:0)+(h&7)+(h>=8?16:0).
__global__ void pack_b(const float* __restrict__ W, unsigned short* __restrict__ dst,
                       int Kin, int kch) {
  int idx = blockIdx.x * blockDim.x + threadIdx.x;
  int total = 16 * kch * 512;
  if (idx >= total) return;
  int h    = idx & 15;
  int lane = (idx >> 4) & 31;
  int rest = idx >> 9;
  int kc   = rest % kch;
  int tile = rest / kch;
  int N = tile * 16 + (lane & 15);
  int K = kc * 32 + ((lane >> 4) << 3) + (h & 7) + ((h >> 3) << 4);
  float v = (K < Kin) ? W[N * Kin + K] : 0.0f;
  dst[idx] = f2bfbits(v);
}

struct DirParams {
  const unsigned short* Bx;   // packed input-weight fragments  [16][KX][512]
  const unsigned short* Bh;   // packed recurrent fragments      [16][2][512]
  const float* bias;          // [256] (PyTorch gate order i,f,g,o)
  int t0, dt, count, dirofs, lastofs;
};

// KX = input k-chunks (2: K<=64 padded, 4: K=128).
// L0: input fp32 [B][T][40], else bf16 [B][T][128].
// BLDS: stage weights in LDS (for KX=4, keeps VGPRs < 256) vs persistent VGPRs.
template<int KX, bool L0, bool BLDS>
__global__ __launch_bounds__(128)
void lstm_kernel(const float* __restrict__ x32, const unsigned short* __restrict__ xbf,
                 DirParams pf, DirParams pb,
                 unsigned short* __restrict__ hout,  // bf16 [B][T][128] (or null)
                 float* __restrict__ hlast,          // f32  [B][128]    (or null)
                 int T) {
  const DirParams p = (blockIdx.y == 0) ? pf : pb;
  const int tid  = threadIdx.x;          // 0..127 (4 waves)
  const int m0   = blockIdx.x * 16;
  const int w    = tid >> 5;             // wave id = unit range [16w,16w+16)
  const int lane = tid & 31;

  __shared__ alignas(16) unsigned short xtile[16 * KX * 32];
  __shared__ alignas(16) unsigned short htile[16 * 64];
  __shared__ alignas(16) unsigned short bsm[BLDS ? (16 * (KX + 2) * 512) : 16];

  for (int i = tid; i < 16 * KX * 32; i += 128) xtile[i] = 0;   // K padding stays 0
  for (int i = tid; i < 16 * 64;      i += 128) htile[i] = 0;   // h0 = 0

  // ---- weight residency ----
  v16bf bxr[4][KX], bhr[4][2];
  if constexpr (BLDS) {
    // stage the full packed Bx|Bh into LDS once (uint4 copies)
    const uint4* sx = (const uint4*)p.Bx;      // 16*KX*512/8 = KX*1024 uint4
    uint4* dx = (uint4*)bsm;
    for (int i = tid; i < KX * 1024; i += 128) dx[i] = sx[i];
    const uint4* sh = (const uint4*)p.Bh;      // 16*2*512/8 = 2048 uint4
    uint4* dh = (uint4*)(bsm + 16 * KX * 512);
    for (int i = tid; i < 2048; i += 128) dh[i] = sh[i];
  } else {
#pragma unroll
    for (int q = 0; q < 4; ++q) {
#pragma unroll
      for (int kc = 0; kc < KX; ++kc)
        bxr[q][kc] = load_B_frag(p.Bx + ((4 * q + w) * KX + kc) * 512);
#pragma unroll
      for (int kc = 0; kc < 2; ++kc)
        bhr[q][kc] = load_B_frag(p.Bh + ((4 * q + w) * 2 + kc) * 512);
    }
  }

  // per-lane gate bias + persistent cell state (rows Mb..Mb+7, unit)
  const int unit = w * 16 + (lane & 15);           // 0..63
  const int Mb   = (lane >> 4) << 3;               // C/D layout: M = r + 8*(lane>=16)
  const float b_i = p.bias[unit],       b_f = p.bias[64 + unit];
  const float b_g = p.bias[128 + unit], b_o = p.bias[192 + unit];
  v8f cst = {};

  __syncthreads();

  for (int it = 0; it < p.count; ++it) {
    const int t = p.t0 + it * p.dt;

    // ---- phase 1: stage x_t tile in LDS (bf16, A-layout row-major) ----
    if constexpr (L0) {
      const int Din = 40;
      for (int i = tid; i < 16 * Din; i += 128) {
        int r = i / Din, k = i - r * Din;
        xtile[r * (KX * 32) + k] =
            f2bfbits(x32[((size_t)(m0 + r) * T + t) * Din + k]);
      }
    } else {
      uint4* dst = (uint4*)xtile;                  // 16 rows x 16 uint4
      for (int i = tid; i < 256; i += 128) {
        int r = i >> 4, j = i & 15;
        dst[i] = ((const uint4*)(xbf + ((size_t)(m0 + r) * T + t) * 128))[j];
      }
    }
    // prefetch next timestep's rows (global_prefetch_b8)
    if (tid < 16 && it + 1 < p.count) {
      int tn = t + p.dt;
      if (L0) __builtin_prefetch(x32 + ((size_t)(m0 + tid) * T + tn) * 40, 0, 1);
      else    __builtin_prefetch(xbf + ((size_t)(m0 + tid) * T + tn) * 128, 0, 1);
    }
    __syncthreads();   // (A) xtile ready; prev-step htile writes visible

    // ---- phase 2: 4-quadrant WMMA, A reused across gates ----
    v8f acc[4] = {};
#pragma unroll
    for (int kc = 0; kc < KX; ++kc) {
      v16bf a = load_A_frag(xtile, KX * 32, kc);
#pragma unroll
      for (int q = 0; q < 4; ++q) {
        v16bf b;
        if constexpr (BLDS) b = load_B_frag(bsm + ((4 * q + w) * KX + kc) * 512);
        else                b = bxr[q][kc];
        acc[q] = __builtin_amdgcn_wmma_f32_16x16x32_bf16(false, a, false, b,
                                                         (short)0, acc[q], false, false);
      }
    }
#pragma unroll
    for (int kc = 0; kc < 2; ++kc) {
      v16bf a = load_A_frag(htile, 64, kc);
#pragma unroll
      for (int q = 0; q < 4; ++q) {
        v16bf b;
        if constexpr (BLDS) b = load_B_frag(bsm + 16 * KX * 512 + ((4 * q + w) * 2 + kc) * 512);
        else                b = bhr[q][kc];
        acc[q] = __builtin_amdgcn_wmma_f32_16x16x32_bf16(false, a, false, b,
                                                         (short)0, acc[q], false, false);
      }
    }
    __syncthreads();   // (B) all htile/xtile reads complete before updates

    // ---- phase 3: in-register LSTM cell update ----
    const bool wlast = (hlast != nullptr) && (it == p.count - 1);
#pragma unroll
    for (int r = 0; r < 8; ++r) {
      float iv = sigmoidf_(acc[0][r] + b_i);
      float fv = sigmoidf_(acc[1][r] + b_f);
      float gv = tanhf    (acc[2][r] + b_g);
      float ov = sigmoidf_(acc[3][r] + b_o);
      float cn = fv * cst[r] + iv * gv;
      cst[r] = cn;
      float hv = ov * tanhf(cn);
      htile[(Mb + r) * 64 + unit] = f2bfbits(hv);
      if (hout)
        hout[((size_t)(m0 + Mb + r) * T + t) * 128 + p.dirofs + unit] = f2bfbits(hv);
      if (wlast)
        hlast[(size_t)(m0 + Mb + r) * 128 + p.lastofs + unit] = hv;
    }
  }
}

// out[B][256] = last[B][128] @ w_fc^T + b_fc   (WMMA, K=128 -> 4 chunks)
__global__ __launch_bounds__(512)
void fc_kernel(const float* __restrict__ last, const unsigned short* __restrict__ Bfc,
               const float* __restrict__ bias, float* __restrict__ out) {
  __shared__ alignas(16) unsigned short atile[16 * 128];
  const int tid = threadIdx.x;
  const int m0  = blockIdx.x * 16;
  for (int i = tid; i < 2048; i += 512)
    atile[i] = f2bfbits(last[(size_t)m0 * 128 + i]);
  __syncthreads();

  const int w = tid >> 5, lane = tid & 31;
  v8f acc = {};
#pragma unroll
  for (int kc = 0; kc < 4; ++kc) {
    v16bf a = load_A_frag(atile, 128, kc);
    v16bf b = load_B_frag(Bfc + (w * 4 + kc) * 512);
    acc = __builtin_amdgcn_wmma_f32_16x16x32_bf16(false, a, false, b,
                                                  (short)0, acc, false, false);
  }
  const int N  = w * 16 + (lane & 15);
  const int Mb = (lane >> 4) << 3;
  const float bv = bias[N];
#pragma unroll
  for (int r = 0; r < 8; ++r)
    out[(size_t)(m0 + Mb + r) * 256 + N] = acc[r] + bv;
}

extern "C" void kernel_launch(void* const* d_in, const int* in_sizes, int n_in,
                              void* d_out, int out_size, void* d_ws, size_t ws_size,
                              hipStream_t stream) {
  const int T = 250, D = 40;
  const float* x     = (const float*)d_in[0];
  const float* wih0f = (const float*)d_in[1];
  const float* whh0f = (const float*)d_in[2];
  const float* b0f   = (const float*)d_in[3];
  const float* wih0b = (const float*)d_in[4];
  const float* whh0b = (const float*)d_in[5];
  const float* b0b   = (const float*)d_in[6];
  const float* wih1f = (const float*)d_in[7];
  const float* whh1f = (const float*)d_in[8];
  const float* b1f   = (const float*)d_in[9];
  const float* wih1b = (const float*)d_in[10];
  const float* whh1b = (const float*)d_in[11];
  const float* b1b   = (const float*)d_in[12];
  const float* wfc   = (const float*)d_in[13];
  const float* bfc   = (const float*)d_in[14];
  float* out = (float*)d_out;
  const int B = in_sizes[0] / (T * D);     // 4096

  char* wsb = (char*)d_ws;
  size_t off = 0;
  auto alloc = [&](size_t bytes) -> void* {
    void* p = wsb + off;
    off = (off + bytes + 255) & ~(size_t)255;
    return p;
  };
  const size_t FR2 = (size_t)16 * 2 * 512 * 2;   // K<=64 packed bytes
  const size_t FR4 = (size_t)16 * 4 * 512 * 2;   // K=128 packed bytes
  unsigned short* Bx0f = (unsigned short*)alloc(FR2);
  unsigned short* Bh0f = (unsigned short*)alloc(FR2);
  unsigned short* Bx0b = (unsigned short*)alloc(FR2);
  unsigned short* Bh0b = (unsigned short*)alloc(FR2);
  unsigned short* Bx1f = (unsigned short*)alloc(FR4);
  unsigned short* Bh1f = (unsigned short*)alloc(FR2);
  unsigned short* Bx1b = (unsigned short*)alloc(FR4);
  unsigned short* Bh1b = (unsigned short*)alloc(FR2);
  unsigned short* Bfc  = (unsigned short*)alloc(FR4);
  unsigned short* h1   = (unsigned short*)alloc((size_t)B * T * 128 * 2);
  float*          last = (float*)alloc((size_t)B * 128 * 4);

  auto pack = [&](const float* W, unsigned short* dst, int Kin, int kch) {
    int total = 16 * kch * 512;
    pack_b<<<(total + 255) / 256, 256, 0, stream>>>(W, dst, Kin, kch);
  };
  pack(wih0f, Bx0f, 40, 2);  pack(whh0f, Bh0f, 64, 2);
  pack(wih0b, Bx0b, 40, 2);  pack(whh0b, Bh0b, 64, 2);
  pack(wih1f, Bx1f, 128, 4); pack(whh1f, Bh1f, 64, 2);
  pack(wih1b, Bx1b, 128, 4); pack(whh1b, Bh1b, 64, 2);
  pack(wfc,   Bfc,  128, 4);

  // Layer 0: both directions in one launch (grid.y), full 250-step chains.
  // Weights persistent in VGPRs (KX=2 fits comfortably under 256 VGPRs).
  DirParams p0{Bx0f, Bh0f, b0f, 0,     +1, T, 0,  0};
  DirParams p1{Bx0b, Bh0b, b0b, T - 1, -1, T, 64, 0};
  lstm_kernel<2, true, false><<<dim3(B / 16, 2), 128, 0, stream>>>(
      x, nullptr, p0, p1, h1, nullptr, T);

  // Layer 1: fwd needs the full chain (last h at t=249); bwd half of
  // h2[:, -1, :] is the reverse scan's FIRST step -> a single step at t=249.
  // Weights staged once in LDS (KX=4 would exceed the 256-VGPR window).
  DirParams q0{Bx1f, Bh1f, b1f, 0,     +1, T, 0, 0};
  DirParams q1{Bx1b, Bh1b, b1b, T - 1, -1, 1, 0, 64};
  lstm_kernel<4, false, true><<<dim3(B / 16, 2), 128, 0, stream>>>(
      nullptr, h1, q0, q1, nullptr, last, T);

  fc_kernel<<<B / 16, 512, 0, stream>>>(last, Bfc, bfc, out);
}
